// WeightFP8ActFP8StaticQuantLinear_71794673320598
// MI455X (gfx1250) — 8-bit matrix operand order repaired and checked by reading, compiled, not run on hardware
//
#include <hip/hip_runtime.h>
#include <hip/hip_bf16.h>
#include <stdint.h>
#include <stddef.h>

typedef __attribute__((ext_vector_type(16))) int   v16i;
typedef __attribute__((ext_vector_type(8)))  float v8f;

#define FP8_MAX_V 448.0f
#define TILE_M 128
#define TILE_N 128
#define TILE_K 128
#define LDS_STRIDE 144
#define TILE_BYTES (TILE_M * LDS_STRIDE)

typedef int v4i_g __attribute__((vector_size(16)));
typedef __attribute__((address_space(1))) v4i_g gv4i;
typedef __attribute__((address_space(3))) v4i_g lv4i;

static __device__ __forceinline__ gv4i* as_global_v4(const void* p) {
    return (gv4i*)(unsigned long long)(uintptr_t)p;
}
static __device__ __forceinline__ lv4i* as_lds_v4(const void* p) {
    return (lv4i*)(unsigned int)(uintptr_t)p;
}

#if __has_builtin(__builtin_amdgcn_global_load_async_to_lds_b128)
#define HAS_ASYNC_LDS 1
#else
#define HAS_ASYNC_LDS 0
#endif

static __device__ __forceinline__ void wait_async0() {
#if __has_builtin(__builtin_amdgcn_s_wait_asynccnt)
    __builtin_amdgcn_s_wait_asynccnt(0);
#else
    asm volatile("s_wait_asynccnt 0x0" ::: "memory");
#endif
}

__global__ void quant_act_kernel(const __hip_bfloat16* __restrict__ x,
                                 const __hip_bfloat16* __restrict__ iscale,
                                 uint32_t* __restrict__ out, int ndw) {
    int i = blockIdx.x * blockDim.x + threadIdx.x;
    if (i >= ndw) return;
    float rs = 1.0f / __bfloat162float(iscale[0]);
    const __hip_bfloat16* p = x + (size_t)i * 4;
    float f0 = fminf(fmaxf(__bfloat162float(p[0]) * rs, -FP8_MAX_V), FP8_MAX_V);
    float f1 = fminf(fmaxf(__bfloat162float(p[1]) * rs, -FP8_MAX_V), FP8_MAX_V);
    float f2 = fminf(fmaxf(__bfloat162float(p[2]) * rs, -FP8_MAX_V), FP8_MAX_V);
    float f3 = fminf(fmaxf(__bfloat162float(p[3]) * rs, -FP8_MAX_V), FP8_MAX_V);
    int packed = __builtin_amdgcn_cvt_pk_fp8_f32(f0, f1, 0, false);
    packed     = __builtin_amdgcn_cvt_pk_fp8_f32(f2, f3, packed, true);
    out[i] = (uint32_t)packed;
}

__global__ void quant_w_kernel(const __hip_bfloat16* __restrict__ w,
                               uint32_t* __restrict__ out, int ndw) {
    int i = blockIdx.x * blockDim.x + threadIdx.x;
    if (i >= ndw) return;
    const __hip_bfloat16* p = w + (size_t)i * 4;
    int packed = __builtin_amdgcn_cvt_pk_fp8_f32(__bfloat162float(p[0]),
                                                 __bfloat162float(p[1]), 0, false);
    packed     = __builtin_amdgcn_cvt_pk_fp8_f32(__bfloat162float(p[2]),
                                                 __bfloat162float(p[3]), packed, true);
    out[i] = (uint32_t)packed;
}

__global__ __launch_bounds__(256)
void gemm_fp8_kernel(const uint8_t* __restrict__ A8,
                     const uint8_t* __restrict__ W8,
                     const __hip_bfloat16* __restrict__ wscale,
                     const __hip_bfloat16* __restrict__ iscale,
                     const __hip_bfloat16* __restrict__ bias,
                     __hip_bfloat16* __restrict__ out,
                     int M, int N, int K) {
    __shared__ __align__(16) uint8_t smemA[2 * TILE_BYTES];
    __shared__ __align__(16) uint8_t smemW[2 * TILE_BYTES];

    const int tid     = threadIdx.x;
    const int wave    = tid >> 5;
    const int lane    = tid & 31;
    const int laneRow = lane & 15;
    const int hi      = lane >> 4;

    const int wm = wave >> 2;
    const int wn = wave & 3;

    const int gm0 = blockIdx.y * TILE_M;
    const int gn0 = blockIdx.x * TILE_N;

    v8f acc[4][2];
#pragma unroll
    for (int mt = 0; mt < 4; ++mt)
#pragma unroll
        for (int nt = 0; nt < 2; ++nt)
#pragma unroll
            for (int j = 0; j < 8; ++j) acc[mt][nt][j] = 0.0f;

    const int nk = K / TILE_K;

    auto stage = [&](int k0, int buf) {
        uint8_t* ldsA = smemA + buf * TILE_BYTES;
        uint8_t* ldsW = smemW + buf * TILE_BYTES;
#pragma unroll
        for (int p = 0; p < 4; ++p) {
            int idx = p * 256 + tid;
            int row = idx >> 3;
            int col = (idx & 7) * 16;
            const uint8_t* ga = A8 + (size_t)(gm0 + row) * (size_t)K + (size_t)(k0 + col);
            const uint8_t* gw = W8 + (size_t)(gn0 + row) * (size_t)K + (size_t)(k0 + col);
            uint8_t* la = ldsA + row * LDS_STRIDE + col;
            uint8_t* lw = ldsW + row * LDS_STRIDE + col;
#if HAS_ASYNC_LDS
            __builtin_amdgcn_global_load_async_to_lds_b128(as_global_v4(ga), as_lds_v4(la), 0, 0);
            __builtin_amdgcn_global_load_async_to_lds_b128(as_global_v4(gw), as_lds_v4(lw), 0, 0);
#else
            *(uint4*)la = *(const uint4*)ga;
            *(uint4*)lw = *(const uint4*)gw;
#endif
        }
    };

    stage(0, 0);

    for (int it = 0; it < nk; ++it) {
        const int cur = it & 1;
        wait_async0();
        __syncthreads();
        if (it + 1 < nk)
            stage((it + 1) * TILE_K, 1 - cur);

        const uint8_t* ldsA = smemA + cur * TILE_BYTES;
        const uint8_t* ldsW = smemW + cur * TILE_BYTES;

        v16i bfrag[2];
#pragma unroll
        for (int nt = 0; nt < 2; ++nt) {
            const uint8_t* brow = ldsW + (wn * 32 + nt * 16 + laneRow) * LDS_STRIDE + hi * 16;
#pragma unroll
            for (int q = 0; q < 4; ++q) {
                uint4 d = *(const uint4*)(brow + q * 32);
                bfrag[nt][4 * q + 0] = (int)d.x;
                bfrag[nt][4 * q + 1] = (int)d.y;
                bfrag[nt][4 * q + 2] = (int)d.z;
                bfrag[nt][4 * q + 3] = (int)d.w;
            }
        }

        v16i afrag[4];
#pragma unroll
        for (int mt = 0; mt < 4; ++mt) {
            const uint8_t* arow = ldsA + (wm * 64 + mt * 16 + laneRow) * LDS_STRIDE + hi * 16;
#pragma unroll
            for (int p = 0; p < 8; ++p) {
                uint2 d = *(const uint2*)(arow + (p >> 1) * 32 + (p & 1) * 8);
                afrag[mt][2 * p + 0] = (int)d.x;
                afrag[mt][2 * p + 1] = (int)d.y;
            }
        }

#pragma unroll
        for (int mt = 0; mt < 4; ++mt)
#pragma unroll
            for (int nt = 0; nt < 2; ++nt)
                acc[mt][nt] = __builtin_amdgcn_wmma_f32_16x16x128_fp8_fp8(
                    afrag[mt], bfrag[nt], (short)0, acc[mt][nt],
                     false,  false);
    }

    float is = __bfloat162float(iscale[0]);
#pragma unroll
    for (int nt = 0; nt < 2; ++nt) {
        int n = gn0 + wn * 32 + nt * 16 + laneRow;
        float s  = is * __bfloat162float(wscale[n]);
        float bb = __bfloat162float(bias[n]);
#pragma unroll
        for (int mt = 0; mt < 4; ++mt) {
            int mbase = gm0 + wm * 64 + mt * 16 + hi * 8;
#pragma unroll
            for (int j = 0; j < 8; ++j) {
                out[(size_t)(mbase + j) * (size_t)N + (size_t)n] =
                    __float2bfloat16(acc[mt][nt][j] * s + bb);
            }
        }
    }
}

extern "C" void kernel_launch(void* const* d_in, const int* in_sizes, int n_in,
                              void* d_out, int out_size, void* d_ws, size_t ws_size,
                              hipStream_t stream) {
    const __hip_bfloat16* x      = (const __hip_bfloat16*)d_in[0];
    const __hip_bfloat16* w      = (const __hip_bfloat16*)d_in[1];
    const __hip_bfloat16* wscale = (const __hip_bfloat16*)d_in[2];
    const __hip_bfloat16* iscale = (const __hip_bfloat16*)d_in[3];
    const __hip_bfloat16* bias   = (const __hip_bfloat16*)d_in[4];
    __hip_bfloat16*       out    = (__hip_bfloat16*)d_out;

    const int OUT = in_sizes[4];
    const int IN  = in_sizes[1] / OUT;
    const int M   = in_sizes[0] / IN;

    uint8_t* A8 = (uint8_t*)d_ws;
    uint8_t* W8 = A8 + (size_t)M * (size_t)IN;

    const int ndwA = (int)(((size_t)M * IN) / 4);
    const int ndwW = (int)(((size_t)OUT * IN) / 4);
    quant_act_kernel<<<(ndwA + 255) / 256, 256, 0, stream>>>(x, iscale, (uint32_t*)A8, ndwA);
    quant_w_kernel<<<(ndwW + 255) / 256, 256, 0, stream>>>(w, (uint32_t*)W8, ndwW);

    dim3 grid(OUT / TILE_N, M / TILE_M);
    gemm_fp8_kernel<<<grid, 256, 0, stream>>>(A8, W8, wscale, iscale, bias, out, M, OUT, IN);
}
